// Interaction_31190052503577
// MI455X (gfx1250) — compile-verified, matrix operand order checked by reading, not run on hardware
//
#include <hip/hip_runtime.h>
#include <hip/hip_bf16.h>
#include <stddef.h>

// ---------------- problem constants ----------------
#define EN   100000      // edges
#define TN   1000000     // triplets
#define HC   128         // hidden channels
#define BQ   8           // num_bilinear
#define SRN  42          // num_spherical * num_radial
#define RN   6           // num_radial
#define KK   (BQ * HC)   // 1024 : flattened contraction dim of bilinear GEMM
#define MT   4           // M-tiles (of 16 triplets) per workgroup -> 64 triplets/WG
#define MROWS (MT * 16)  // 64

// dynamic LDS layout (bytes)
#define AV_BYTES   (MROWS * KK * 2)        // 131072  A-tile bf16
#define GAT_BYTES  (MROWS * HC * 2)        //  16384  gathered x_kj rows bf16
#define STILE_BYTES (MROWS * BQ * 4)       //   2048  sbf_h tile f32
#define SIDX_BYTES (MROWS * 4)             //    256  idx_ji tile
#define SMEM_BYTES (AV_BYTES + GAT_BYTES + STILE_BYTES + SIDX_BYTES)  // 149760

// ---------------- WMMA vector types ----------------
typedef __attribute__((ext_vector_type(16))) __bf16 v16bf;
typedef __attribute__((ext_vector_type(8)))  float  v8f;
typedef __attribute__((ext_vector_type(4)))  unsigned int uint4v;

union FragBF {
    uint4v u[2];   // two contiguous 16B chunks
    v16bf  v;      // wmma operand
};

// ---------------- bf16 helpers (RNE) ----------------
__device__ __forceinline__ unsigned short f2bf(float f) {
    unsigned int u = __float_as_uint(f);
    u = (u + 0x7FFFu + ((u >> 16) & 1u)) >> 16;
    return (unsigned short)u;
}
__device__ __forceinline__ float bf2f(unsigned short h) {
    return __uint_as_float(((unsigned int)h) << 16);
}
__device__ __forceinline__ float silu(float z) {
    return z * (1.0f / (1.0f + __expf(-z)));
}

// ---------------- kernel 1: W [H,B,H] f32 -> bf16 (layout preserved: [i][j*H+l]) --
__global__ void wconv_kernel(const float* __restrict__ W,
                             unsigned short* __restrict__ Wbf, int n) {
    int i = blockIdx.x * blockDim.x + threadIdx.x;
    if (i < n) Wbf[i] = f2bf(W[i]);
}

// ---------------- kernel 2: fused edge prep -------------------------------------
// x_ji = silu(x@W_ji+b_ji);  x_kj_bf = bf16( silu(x@W_kj+b_kj) * (rbf@W_rbf) )
__global__ __launch_bounds__(HC) void prep_kernel(
    const float* __restrict__ x,     const float* __restrict__ rbf,
    const float* __restrict__ W_rbf, const float* __restrict__ W_kj,
    const float* __restrict__ b_kj,  const float* __restrict__ W_ji,
    const float* __restrict__ b_ji,
    unsigned short* __restrict__ xkj_bf, float* __restrict__ x_ji_out) {
    __shared__ float xs[HC];
    const int e = blockIdx.x;
    const int h = threadIdx.x;
    xs[h] = x[(size_t)e * HC + h];
    __syncthreads();
    float akj = b_kj[h];
    float aji = b_ji[h];
#pragma unroll 8
    for (int l = 0; l < HC; ++l) {
        const float xv = xs[l];
        akj = fmaf(xv, W_kj[l * HC + h], akj);
        aji = fmaf(xv, W_ji[l * HC + h], aji);
    }
    float rh = 0.0f;
#pragma unroll
    for (int r = 0; r < RN; ++r)
        rh = fmaf(rbf[(size_t)e * RN + r], W_rbf[r * HC + h], rh);
    x_ji_out[(size_t)e * HC + h] = silu(aji);
    xkj_bf[(size_t)e * HC + h]   = f2bf(silu(akj) * rh);
}

// ---------------- kernel 3: sbf_h = sbf @ W_sbf  [T,42]x[42,8] ------------------
__global__ void sbf_kernel(const float* __restrict__ sbf,
                           const float* __restrict__ W_sbf,
                           float* __restrict__ sbf_h) {
    const int idx = blockIdx.x * blockDim.x + threadIdx.x;
    if (idx >= TN * BQ) return;
    const int w = idx >> 3;
    const int b = idx & 7;
    const float* row = sbf + (size_t)w * SRN;
    float acc = 0.0f;
#pragma unroll
    for (int j = 0; j < SRN; ++j) acc = fmaf(row[j], W_sbf[j * BQ + b], acc);
    sbf_h[idx] = acc;
}

// ---------------- kernel 4: zero the scatter accumulator ------------------------
__global__ void zero_kernel(float* __restrict__ p, int n) {
    int i = blockIdx.x * blockDim.x + threadIdx.x;
    if (i < n) p[i] = 0.0f;
}

// ---------------- kernel 5: bilinear einsum as bf16 WMMA GEMM + atomic scatter --
// 64 triplets per workgroup (4 M-tiles), 8 wave32s, each wave owns one 16-wide
// N-slice and all 4 M-tiles (register blocking on M => 4x B-operand reuse).
//   A[m][j*128+l] = sbf_h[t0+m, j] * x_kj[idx_kj[t0+m], l]   (bf16, in LDS)
//   C(64x128)     = A(64x1024) @ Wbf(1024x128)               (128 wmma / wave)
//   agg[idx_ji[t0+m], :] += C[m, :]                          (f32 atomics)
__global__ __launch_bounds__(256) void bilinear_kernel(
    const unsigned short* __restrict__ xkj_bf,   // [E][128] bf16
    const float*          __restrict__ sbf_h,    // [T][8]
    const unsigned short* __restrict__ Wbf,      // [128][1024] bf16, row i = W[i,:,:]
    const int*            __restrict__ idx_kj,
    const int*            __restrict__ idx_ji,
    float*                __restrict__ agg) {    // [E][128]
    extern __shared__ unsigned char smem_raw[];
    unsigned short* Av    = (unsigned short*)smem_raw;                      // [64][1024]
    unsigned short* gat   = (unsigned short*)(smem_raw + AV_BYTES);         // [64][128]
    float*          stile = (float*)(smem_raw + AV_BYTES + GAT_BYTES);      // [64][8]
    int*            sidx  = (int*)(smem_raw + AV_BYTES + GAT_BYTES + STILE_BYTES); // [64]

    const int t0   = blockIdx.x * MROWS;
    const int tid  = threadIdx.x;        // 0..255
    const int wave = tid >> 5;           // 0..7  -> owns N-slice [wave*16, +16)
    const int lane = tid & 31;

    // ---- stage gather + sbf tile + idx_ji (coalesced) ----
    {
        const int m   = tid >> 2;        // 0..63 triplet row
        const int seg = tid & 3;         // 0..3  -> 32 bf16 each
        const int e   = idx_kj[t0 + m];
        const uint4v* src = (const uint4v*)(xkj_bf + (size_t)e * HC + seg * 32);
        uint4v* dst = (uint4v*)(gat + m * HC + seg * 32);
#pragma unroll
        for (int q = 0; q < 4; ++q) dst[q] = src[q];
        // 512 sbf values: 2 per thread
        stile[tid]       = sbf_h[(size_t)t0 * BQ + tid];
        stile[tid + 256] = sbf_h[(size_t)t0 * BQ + tid + 256];
        if (tid < MROWS) sidx[tid] = idx_ji[t0 + tid];
    }
    __syncthreads();

    // ---- build A-tile: thread -> (row m, quarter q = two j's x 128 l) ----
    {
        const int m = tid >> 2;
        const int q = tid & 3;
#pragma unroll
        for (int jj = 0; jj < 2; ++jj) {
            const int j = q * 2 + jj;
            const float s = stile[m * BQ + j];
            unsigned short* arow = Av + m * KK + j * HC;
            const unsigned short* vrow = gat + m * HC;
#pragma unroll 8
            for (int l = 0; l < HC; ++l)
                arow[l] = f2bf(s * bf2f(vrow[l]));
        }
    }
    __syncthreads();

    // ---- WMMA main loop: K = 1024 in 32 steps of 32, 4 M-tiles per step ----
    const int nbase = wave * 16;
    const int mrow  = lane & 15;              // row within M-tile (lanes 16-31 mirror)
    const int ncol  = lane & 15;              // B column
    const int hi    = lane >> 4;              // K half-select per ISA layout
    const unsigned short* wrow = Wbf + (size_t)(nbase + ncol) * KK;

    v8f acc[MT] = {};
#pragma unroll 2
    for (int kc = 0; kc < 32; ++kc) {
        const int kb = kc * 32 + hi * 8;      // K = {kb..kb+7, kb+16..kb+23}
        FragBF b;
        b.u[0] = *(const uint4v*)(wrow + kb);
        b.u[1] = *(const uint4v*)(wrow + kb + 16);
#pragma unroll
        for (int mt = 0; mt < MT; ++mt) {
            FragBF a;
            const unsigned short* arow = Av + (mt * 16 + mrow) * KK + kb;
            a.u[0] = *(const uint4v*)(arow);
            a.u[1] = *(const uint4v*)(arow + 16);
            acc[mt] = __builtin_amdgcn_wmma_f32_16x16x32_bf16(
                          false, a.v, false, b.v, (short)0, acc[mt], false, false);
        }
    }

    // ---- scatter C-tiles with f32 atomics (lane n, vgpr r -> M=r(+8), N=n) ----
#pragma unroll
    for (int mt = 0; mt < MT; ++mt) {
#pragma unroll
        for (int r = 0; r < 8; ++r) {
            const int m = mt * 16 + (hi ? (r + 8) : r);
            const int e = sidx[m];
            atomicAdd(&agg[(size_t)e * HC + nbase + ncol], acc[mt][r]);
        }
    }
}

// ---------------- kernel 6: h = silu((x_ji + agg) @ W_lin + b_lin) --------------
__global__ __launch_bounds__(HC) void final_kernel(
    const float* __restrict__ x_ji, const float* __restrict__ agg,
    const float* __restrict__ W_lin, const float* __restrict__ b_lin,
    float* __restrict__ out) {
    __shared__ float xs[HC];
    const int e = blockIdx.x;
    const int h = threadIdx.x;
    xs[h] = x_ji[(size_t)e * HC + h] + agg[(size_t)e * HC + h];
    __syncthreads();
    float a = b_lin[h];
#pragma unroll 8
    for (int l = 0; l < HC; ++l) a = fmaf(xs[l], W_lin[l * HC + h], a);
    out[(size_t)e * HC + h] = silu(a);
}

// ---------------- launcher ------------------------------------------------------
extern "C" void kernel_launch(void* const* d_in, const int* in_sizes, int n_in,
                              void* d_out, int out_size, void* d_ws, size_t ws_size,
                              hipStream_t stream) {
    // setup_inputs() dict order
    const float* x      = (const float*)d_in[0];
    const float* rbf    = (const float*)d_in[1];
    const float* sbf    = (const float*)d_in[2];
    const int*   idx_kj = (const int*)d_in[3];
    const int*   idx_ji = (const int*)d_in[4];
    const float* W_rbf  = (const float*)d_in[5];
    const float* W_sbf  = (const float*)d_in[6];
    const float* W_kj   = (const float*)d_in[7];
    const float* b_kj   = (const float*)d_in[8];
    const float* W_ji   = (const float*)d_in[9];
    const float* b_ji   = (const float*)d_in[10];
    const float* W      = (const float*)d_in[11];
    const float* W_lin  = (const float*)d_in[12];
    const float* b_lin  = (const float*)d_in[13];
    float* out = (float*)d_out;

    // workspace carve-up (~160.3 MB total), 256B aligned
    char* ws = (char*)d_ws;
    size_t off = 0;
    auto take = [&](size_t bytes) -> char* {
        char* p = ws + off;
        off = (off + bytes + 255) & ~(size_t)255;
        return p;
    };
    unsigned short* xkj_bf = (unsigned short*)take((size_t)EN * HC * 2); // 25.6 MB
    float*          x_ji   = (float*)take((size_t)EN * HC * 4);          // 51.2 MB
    float*          sbf_h  = (float*)take((size_t)TN * BQ * 4);          // 32.0 MB
    unsigned short* Wbf    = (unsigned short*)take((size_t)HC * KK * 2); // 256 KB
    float*          agg    = (float*)take((size_t)EN * HC * 4);          // 51.2 MB
    (void)ws_size;

    // 1) weight -> bf16 (L2-resident for the whole GEMM)
    wconv_kernel<<<(HC * KK + 255) / 256, 256, 0, stream>>>(W, Wbf, HC * KK);
    // 2) fused edge-side prep
    prep_kernel<<<EN, HC, 0, stream>>>(x, rbf, W_rbf, W_kj, b_kj, W_ji, b_ji,
                                       xkj_bf, x_ji);
    // 3) sbf projection
    sbf_kernel<<<(TN * BQ + 255) / 256, 256, 0, stream>>>(sbf, W_sbf, sbf_h);
    // 4) zero scatter target
    zero_kernel<<<(EN * HC + 255) / 256, 256, 0, stream>>>(agg, EN * HC);
    // 5) main bilinear WMMA GEMM + scatter (T/64 = 15625 workgroups, 150KB LDS)
    bilinear_kernel<<<TN / MROWS, 256, SMEM_BYTES, stream>>>(xkj_bf, sbf_h, Wbf,
                                                             idx_kj, idx_ji, agg);
    // 6) output projection + silu
    final_kernel<<<EN, HC, 0, stream>>>(x_ji, agg, W_lin, b_lin, out);
    (void)n_in; (void)in_sizes; (void)out_size;
}
